// RonfiguredAttentionModule_48369921687717
// MI455X (gfx1250) — matrix operand order repaired and checked by reading, not compiled, not run on hardware
//
#include <hip/hip_runtime.h>
#include <hip/hip_bf16.h>

// ---------------- problem constants ----------------
#define B_        4
#define N_        16384          // H*W patch tokens
#define MDET      100
#define NM        16484          // N_ + MDET
#define C_        256
#define C3        768            // 3*C
#define NH        8
#define CH        32
#define HS        128            // H = W = 128
#define ROWS_TOTAL (B_*NM)       // 65936
#define MT_TOTAL   (ROWS_TOTAL/16) // 4121 (exact)

typedef __attribute__((ext_vector_type(16))) _Float16 v16h;
typedef __attribute__((ext_vector_type(8)))  _Float16 v8h;
typedef __attribute__((ext_vector_type(8)))  float    v8f;

__device__ __forceinline__ v8f wmma_f16(v16h a, v16h b, v8f c) {
  // D = A(16x32 f16) * B(32x16 f16) + C(f32)
  return __builtin_amdgcn_wmma_f32_16x16x32_f16(
      /*neg_a=*/false, a, /*neg_b=*/false, b,
      /*c_mod=*/(short)0, c, /*reuse_a=*/false, /*reuse_b=*/false);
}

// A-fragment (16x32 f16) from a contiguous fp32 row of 32 K values.
// Layout: lane row = lane&15, hi = lane>>4;
//   elem j in 0..15 -> K = hi*16 + j
__device__ __forceinline__ v16h load_a32(const float* __restrict__ p, int hi) {
  const float4* q0 = (const float4*)(p + hi * 16);
  const float4* q1 = (const float4*)(p + hi * 16 + 8);
  float4 f0 = q0[0], f1 = q0[1], g0 = q1[0], g1 = q1[1];
  v16h a;
  a[0]=(_Float16)f0.x;  a[1]=(_Float16)f0.y;  a[2]=(_Float16)f0.z;  a[3]=(_Float16)f0.w;
  a[4]=(_Float16)f1.x;  a[5]=(_Float16)f1.y;  a[6]=(_Float16)f1.z;  a[7]=(_Float16)f1.w;
  a[8]=(_Float16)g0.x;  a[9]=(_Float16)g0.y;  a[10]=(_Float16)g0.z; a[11]=(_Float16)g0.w;
  a[12]=(_Float16)g1.x; a[13]=(_Float16)g1.y; a[14]=(_Float16)g1.z; a[15]=(_Float16)g1.w;
  return a;
}

// A-fragment (16x32 f16) directly from a contiguous f16 row of 32 K values:
// two 16-byte vector loads, no conversion VALU.
__device__ __forceinline__ v16h load_a32h(const _Float16* __restrict__ p, int hi) {
  v8h lo = *(const v8h*)(p + hi * 16);
  v8h hh = *(const v8h*)(p + hi * 16 + 8);
  v16h a;
#pragma unroll
  for (int j = 0; j < 8; ++j) { a[j] = lo[j]; a[j + 8] = hh[j]; }
  return a;
}

// =====================================================================
// K1: qkv(f16) = concat(x, det) @ qkv_w + qkv_b   [65936x256]x[256x768]
//   grid (516, 48), block 256 (8 waves): 8 M-tiles x 1 N-tile per block.
//   Weight panel [256 x 16] staged fp32->f16 into LDS, transposed.
// =====================================================================
__global__ __launch_bounds__(256)
void qkv_gemm_kernel(const float* __restrict__ x, const float* __restrict__ det,
                     const float* __restrict__ w, const float* __restrict__ bias,
                     _Float16* __restrict__ out) {
  __shared__ _Float16 BsT[16][C_ + 8];   // [n within tile][K]
  const int t  = threadIdx.x;
  const int n0 = blockIdx.y * 16;
  {
    const float* wr = w + (size_t)t * C3 + n0;   // row t of weights
#pragma unroll
    for (int j = 0; j < 16; ++j) BsT[j][t] = (_Float16)wr[j];
  }
  __syncthreads();

  const int wave = t >> 5, lane = t & 31;
  const int mtile = blockIdx.x * 8 + wave;
  if (mtile >= MT_TOTAL) return;        // wave-uniform: safe around WMMA

  const int hi = lane >> 4, ln = lane & 15;
  const int lrow = mtile * 16 + ln;
  const int b = lrow / NM, i = lrow % NM;
  const float* rowPtr = (i < N_) ? (x   + ((size_t)b * N_   + i)        * C_)
                                 : (det + ((size_t)b * MDET + (i - N_)) * C_);
  v8f acc = {};
#pragma unroll
  for (int kk = 0; kk < C_; kk += 32) {
    v16h a = load_a32(rowPtr + kk, hi);
    v16h bf;
    const _Float16* bp = &BsT[ln][kk + hi * 16];  // lanes 0-15: K 0..15, 16-31: K 16..31
#pragma unroll
    for (int j = 0; j < 16; ++j) bf[j] = bp[j];
    acc = wmma_f16(a, bf, acc);
  }
  const int ncol = n0 + ln;
  const float bv = bias[ncol];
  const size_t base = (size_t)(mtile * 16 + hi * 8) * C3 + ncol;
#pragma unroll
  for (int r = 0; r < 8; ++r) out[base + (size_t)r * C3] = (_Float16)(acc[r] + bv);
}

// =====================================================================
// K2: per-(b,h,c) column softmax stats of pk over 16384 tokens (f16 in).
// =====================================================================
__global__ __launch_bounds__(256)
void ksoft_stats_kernel(const _Float16* __restrict__ qkv,
                        float* __restrict__ kmax, float* __restrict__ ksum) {
  const int idx = blockIdx.x;                   // b*256 + h*32 + c
  const int c = idx & (CH - 1), h = (idx >> 5) & (NH - 1), b = idx >> 8;
  const _Float16* base = qkv + (size_t)b * NM * C3 + C_ + h * CH + c;
  __shared__ float red[256];
  const int t = threadIdx.x;
  float m = -1e30f;
  for (int n = t; n < N_; n += 256) m = fmaxf(m, (float)base[(size_t)n * C3]);
  red[t] = m; __syncthreads();
  for (int s = 128; s > 0; s >>= 1) { if (t < s) red[t] = fmaxf(red[t], red[t + s]); __syncthreads(); }
  const float mx = red[0]; __syncthreads();
  float sum = 0.f;
  for (int n = t; n < N_; n += 256) sum += __expf((float)base[(size_t)n * C3] - mx);
  red[t] = sum; __syncthreads();
  for (int s = 128; s > 0; s >>= 1) { if (t < s) red[t] += red[t + s]; __syncthreads(); }
  if (t == 0) { kmax[idx] = mx; ksum[idx] = red[0]; }
}

// =====================================================================
// K3: factor[b,h] = ksoftmax^T (32x16384) @ V (16384x32), via WMMA.
//   grid 32 blocks (b,h), block 128 = 4 waves, one 16x16 quadrant each.
// =====================================================================
__global__ __launch_bounds__(128)
void factor_kernel(const _Float16* __restrict__ qkv, const float* __restrict__ kmax,
                   const float* __restrict__ ksum, float* __restrict__ factor) {
  const int bh = blockIdx.x, b = bh >> 3, h = bh & 7;
  __shared__ _Float16 kT[CH][32 + 8];   // [channel][token]  (A is ksoft^T)
  __shared__ _Float16 vT[32][CH + 8];   // [token][channel]
  __shared__ float smax[CH], srcp[CH];
  const int t = threadIdx.x;
  if (t < CH) { smax[t] = kmax[bh * CH + t]; srcp[t] = 1.0f / ksum[bh * CH + t]; }
  __syncthreads();

  const int wave = t >> 5, lane = t & 31;
  const int mq = wave >> 1, nq = wave & 1;
  const int hi = lane >> 4, ln = lane & 15;
  const _Float16* kptr = qkv + (size_t)b * NM * C3 + C_     + h * CH;
  const _Float16* vptr = qkv + (size_t)b * NM * C3 + 2 * C_ + h * CH;
  v8f acc = {};
  for (int n0 = 0; n0 < N_; n0 += 32) {
    __syncthreads();
#pragma unroll
    for (int j = 0; j < 8; ++j) {                // stage 32 tokens x 32 channels
      const int idx = t + j * 128;
      const int tok = idx >> 5, ch = idx & 31;
      const size_t off = (size_t)(n0 + tok) * C3 + ch;
      kT[ch][tok] = (_Float16)(__expf((float)kptr[off] - smax[ch]) * srcp[ch]);
      vT[tok][ch] = vptr[off];
    }
    __syncthreads();
    v16h a, bb;
    const int c = mq * 16 + ln;
#pragma unroll
    for (int j = 0; j < 8; ++j) { a[j] = kT[c][hi * 16 + j]; a[j + 8] = kT[c][hi * 16 + 8 + j]; }
    const int d = nq * 16 + ln;
#pragma unroll
    for (int j = 0; j < 16; ++j) bb[j] = vT[hi * 16 + j][d];
    acc = wmma_f16(a, bb, acc);
  }
#pragma unroll
  for (int r = 0; r < 8; ++r) {
    const int row = mq * 16 + hi * 8 + r, col = nq * 16 + ln;
    factor[((size_t)bh * CH + row) * CH + col] = acc[r];
  }
}

// =====================================================================
// K4: CRPE depthwise conv (k=3/5/7 per head group) on V-image, times Q.
//   Writes attn_out (=) for patch rows. grid (64 tiles, 256 ch, 4 b).
// =====================================================================
__global__ __launch_bounds__(256)
void crpe_conv_kernel(const _Float16* __restrict__ qkv,
                      const float* __restrict__ w3, const float* __restrict__ b3,
                      const float* __restrict__ w5, const float* __restrict__ b5,
                      const float* __restrict__ w7, const float* __restrict__ b7,
                      float* __restrict__ attn_out) {
  const int tile = blockIdx.x, hc = blockIdx.y, b = blockIdx.z;
  const int ty0 = (tile >> 3) * 16, tx0 = (tile & 7) * 16;
  const int t = threadIdx.x, lx = t & 15, ly = t >> 4;
  __shared__ float vt[22][24];                  // 16x16 tile + 3-halo (max k=7)

  int ks; const float* wp; float bias;
  if (hc < 64)       { ks = 3; wp = w3 + hc * 9;          bias = b3[hc]; }
  else if (hc < 160) { ks = 5; wp = w5 + (hc - 64) * 25;  bias = b5[hc - 64]; }
  else               { ks = 7; wp = w7 + (hc - 160) * 49; bias = b7[hc - 160]; }
  const int pad = ks >> 1;
  const _Float16* vbase = qkv + (size_t)b * NM * C3 + 2 * C_ + hc;

  for (int idx = t; idx < 22 * 22; idx += 256) {
    const int yy = ty0 - 3 + idx / 22, xx = tx0 - 3 + idx % 22;
    float v = 0.f;
    if (yy >= 0 && yy < HS && xx >= 0 && xx < HS) v = (float)vbase[(size_t)(yy * HS + xx) * C3];
    vt[idx / 22][idx % 22] = v;
  }
  __syncthreads();

  float acc = bias;
  const int cy = ly + 3, cx = lx + 3;
  for (int dy = -pad; dy <= pad; ++dy)
    for (int dx = -pad; dx <= pad; ++dx)
      acc += wp[(dy + pad) * ks + (dx + pad)] * vt[cy + dy][cx + dx];

  const int n = (ty0 + ly) * HS + (tx0 + lx);
  const float q = (float)qkv[((size_t)b * NM + n) * C3 + hc];   // q channel hc
  attn_out[((size_t)b * NM + n) * C_ + hc] = q * acc;
}

// =====================================================================
// K5: attn_out += scale * (Q @ factor), WMMA, K=32 exactly.
//   grid (128 token-blocks, 32 bh), block 256 = 8 waves x 16 tokens.
// =====================================================================
__global__ __launch_bounds__(256)
void factor_att_kernel(const _Float16* __restrict__ qkv, const float* __restrict__ factor,
                       float* __restrict__ attn_out) {
  const int bh = blockIdx.y, b = bh >> 3, h = bh & 7;
  __shared__ _Float16 fLds[CH][CH + 8];         // [c][d]
  const int t = threadIdx.x;
#pragma unroll
  for (int j = 0; j < 4; ++j) {
    const int idx = t + j * 256;
    fLds[idx >> 5][idx & 31] = (_Float16)factor[(size_t)bh * CH * CH + idx];
  }
  __syncthreads();

  const int wave = t >> 5, lane = t & 31, hi = lane >> 4, ln = lane & 15;
  const int tok0 = blockIdx.x * 128 + wave * 16;
  const _Float16* qrow = qkv + ((size_t)b * NM + tok0 + ln) * C3 + h * CH;
  v16h a = load_a32h(qrow, hi);
  const float scale = 0.17677669529663687f;     // 32^-0.5
#pragma unroll
  for (int nq = 0; nq < 2; ++nq) {
    v16h bb;
#pragma unroll
    for (int j = 0; j < 16; ++j) bb[j] = fLds[hi * 16 + j][nq * 16 + ln];
    v8f acc = {};
    acc = wmma_f16(a, bb, acc);
#pragma unroll
    for (int r = 0; r < 8; ++r) {
      const size_t idx = ((size_t)b * NM + tok0 + hi * 8 + r) * C_ + h * CH + nq * 16 + ln;
      attn_out[idx] += scale * acc[r];
    }
  }
}

// =====================================================================
// K6: det branch, standard softmax attention over 100 tokens.
//   grid 32 (b,h), block 128; K/V staged in LDS; 2-pass softmax per row.
// =====================================================================
__global__ __launch_bounds__(128)
void det_attn_kernel(const _Float16* __restrict__ qkv, float* __restrict__ attn_out) {
  const int bh = blockIdx.x, b = bh >> 3, h = bh & 7;
  __shared__ float kL[MDET][CH], vL[MDET][CH];
  const int t = threadIdx.x;
  const _Float16* base = qkv + ((size_t)b * NM + N_) * C3 + h * CH;
  for (int idx = t; idx < MDET * CH; idx += 128) {
    const int r = idx >> 5, c = idx & 31;
    kL[r][c] = (float)base[(size_t)r * C3 + C_ + c];
    vL[r][c] = (float)base[(size_t)r * C3 + 2 * C_ + c];
  }
  __syncthreads();
  if (t >= MDET) return;

  const float scale = 0.17677669529663687f;
  float q[CH];
  const _Float16* qr = base + (size_t)t * C3;
#pragma unroll
  for (int c = 0; c < CH; ++c) q[c] = (float)qr[c] * scale;

  float mx = -1e30f;
  for (int j = 0; j < MDET; ++j) {
    float s = 0.f;
#pragma unroll
    for (int c = 0; c < CH; ++c) s += q[c] * kL[j][c];
    mx = fmaxf(mx, s);
  }
  float sum = 0.f, outv[CH];
#pragma unroll
  for (int c = 0; c < CH; ++c) outv[c] = 0.f;
  for (int j = 0; j < MDET; ++j) {
    float s = 0.f;
#pragma unroll
    for (int c = 0; c < CH; ++c) s += q[c] * kL[j][c];
    const float p = __expf(s - mx);
    sum += p;
#pragma unroll
    for (int c = 0; c < CH; ++c) outv[c] += p * vL[j][c];
  }
  const float rs = 1.0f / sum;
  float* dst = attn_out + ((size_t)b * NM + N_ + t) * C_ + h * CH;
#pragma unroll
  for (int c = 0; c < CH; ++c) dst[c] = outv[c] * rs;
}

// =====================================================================
// K7: out = attn_out @ proj_w + proj_b, with (patch||det) output remap.
//   grid (516, 16), block 256.
// =====================================================================
__global__ __launch_bounds__(256)
void proj_gemm_kernel(const float* __restrict__ attn, const float* __restrict__ w,
                      const float* __restrict__ bias, float* __restrict__ out) {
  __shared__ _Float16 BsT[16][C_ + 8];
  const int t = threadIdx.x;
  const int n0 = blockIdx.y * 16;
  {
    const float* wr = w + (size_t)t * C_ + n0;
#pragma unroll
    for (int j = 0; j < 16; ++j) BsT[j][t] = (_Float16)wr[j];
  }
  __syncthreads();

  const int wave = t >> 5, lane = t & 31;
  const int mtile = blockIdx.x * 8 + wave;
  if (mtile >= MT_TOTAL) return;

  const int hi = lane >> 4, ln = lane & 15;
  const float* rowPtr = attn + (size_t)(mtile * 16 + ln) * C_;
  v8f acc = {};
#pragma unroll
  for (int kk = 0; kk < C_; kk += 32) {
    v16h a = load_a32(rowPtr + kk, hi);
    v16h bf;
    const _Float16* bp = &BsT[ln][kk + hi * 16];
#pragma unroll
    for (int j = 0; j < 16; ++j) bf[j] = bp[j];
    acc = wmma_f16(a, bf, acc);
  }
  const int ncol = n0 + ln;
  const float bv = bias[ncol];
#pragma unroll
  for (int r = 0; r < 8; ++r) {
    const int row = mtile * 16 + hi * 8 + r;
    const int b = row / NM, i = row % NM;
    const size_t dst = (i < N_)
        ? (((size_t)b * N_ + i) * C_ + ncol)                                    // patch part
        : ((size_t)B_ * N_ * C_ + ((size_t)b * MDET + (i - N_)) * C_ + ncol);   // det part
    out[dst] = acc[r] + bv;
  }
}

// =====================================================================
extern "C" void kernel_launch(void* const* d_in, const int* in_sizes, int n_in,
                              void* d_out, int out_size, void* d_ws, size_t ws_size,
                              hipStream_t stream) {
  const float* x      = (const float*)d_in[0];
  const float* det    = (const float*)d_in[1];
  // d_in[2]=H, d_in[3]=W (compile-time constants 128)
  const float* qkv_w  = (const float*)d_in[4];
  const float* qkv_b  = (const float*)d_in[5];
  const float* proj_w = (const float*)d_in[6];
  const float* proj_b = (const float*)d_in[7];
  const float* w3 = (const float*)d_in[8],  *b3 = (const float*)d_in[9];
  const float* w5 = (const float*)d_in[10], *b5 = (const float*)d_in[11];
  const float* w7 = (const float*)d_in[12], *b7 = (const float*)d_in[13];
  float* out = (float*)d_out;

  // workspace layout (f16 qkv first, 8-byte aligned blocks)
  _Float16* qkv = (_Float16*)d_ws;                          // 65936*768 halves (~101 MB)
  float* attn   = (float*)(qkv + (size_t)ROWS_TOTAL * C3);  // 65936*256 floats (~68 MB)
  float* kmax   = attn + (size_t)ROWS_TOTAL * C_;           // 1024
  float* ksum   = kmax + 1024;                              // 1024
  float* factor = ksum + 1024;                              // 32*32*32

  qkv_gemm_kernel   <<<dim3((MT_TOTAL + 7) / 8, C3 / 16), 256, 0, stream>>>(x, det, qkv_w, qkv_b, qkv);
  ksoft_stats_kernel<<<dim3(B_ * NH * CH),                 256, 0, stream>>>(qkv, kmax, ksum);
  factor_kernel     <<<dim3(B_ * NH),                      128, 0, stream>>>(qkv, kmax, ksum, factor);
  crpe_conv_kernel  <<<dim3(64, C_, B_),                   256, 0, stream>>>(qkv, w3, b3, w5, b5, w7, b7, attn);
  factor_att_kernel <<<dim3(N_ / 128, B_ * NH),            256, 0, stream>>>(qkv, factor, attn);
  det_attn_kernel   <<<dim3(B_ * NH),                      128, 0, stream>>>(qkv, attn);
  proj_gemm_kernel  <<<dim3((MT_TOTAL + 7) / 8, C_ / 16),  256, 0, stream>>>(attn, proj_w, proj_b, out);
}
